// ChemSReactMPNN_22694607192499
// MI455X (gfx1250) — compile-verified, matrix operand order checked by reading, not run on hardware
//
#include <hip/hip_runtime.h>
#include <hip/hip_bf16.h>
#include <math.h>

// ---------------------------------------------------------------------------
// ChemSReactMPNN for MI455X (gfx1250, wave32, WMMA).
// Dense GEMMs -> v_wmma_f32_16x16x32_f16, strip-mined over N so each wave
// loads/converts its A fragment once per K-step and reuses it for NT column
// tiles (cuts A-side L2 traffic + f32->f16 conversion work by NT).
// Graph gather/scatter + segment softmax -> global float atomics
// (HBM-bound path, ~5GB total @ 23.3 TB/s dominates runtime).
// ---------------------------------------------------------------------------

typedef __attribute__((ext_vector_type(16))) _Float16 v16h;
typedef __attribute__((ext_vector_type(8)))  float    v8f;

#define NNODES 100000
#define NEDGES 3200000
#define HDIM   64
#define BGRAPH 2048
#define DDIM   192            // 3*H
#define ROUT   1024

__device__ __forceinline__ float sigf(float x) { return 1.0f / (1.0f + __expf(-x)); }

// monotone float<->uint transform for atomic max on signed floats
__device__ __forceinline__ unsigned f2ord(float f) {
  unsigned u = __float_as_uint(f);
  return (u & 0x80000000u) ? ~u : (u | 0x80000000u);
}
__device__ __forceinline__ float ord2f(unsigned u) {
  return __uint_as_float((u & 0x80000000u) ? (u ^ 0x80000000u) : ~u);
}

// ------------------------------ utility ------------------------------------
__global__ void fill_zero_f32(float* p, long long n) {
  long long i = (long long)blockIdx.x * blockDim.x + threadIdx.x;
  if (i < n) p[i] = 0.0f;
}

// ------------------------------ graph ops ----------------------------------
__global__ void deg_kernel(const int* __restrict__ dst, float* __restrict__ deg) {
  int i = blockIdx.x * blockDim.x + threadIdx.x;
  if (i < NEDGES) atomicAdd(&deg[dst[i]], 1.0f);
}

__global__ void recip_deg_kernel(float* deg) {
  int i = blockIdx.x * blockDim.x + threadIdx.x;
  if (i < NNODES) deg[i] = 1.0f / fmaxf(deg[i], 1.0f);
}

// one thread per (edge, feature): agg[dst] += h[src]
__global__ void edge_agg_kernel(const int* __restrict__ src, const int* __restrict__ dst,
                                const float* __restrict__ h, float* __restrict__ agg) {
  long long t = (long long)blockIdx.x * blockDim.x + threadIdx.x;
  if (t >= (long long)NEDGES * HDIM) return;
  int e = (int)(t >> 6);
  int f = (int)(t & 63);
  atomicAdd(&agg[(long long)dst[e] * HDIM + f], h[(long long)src[e] * HDIM + f]);
}

// ------------------------------ WMMA GEMM ----------------------------------
// C[M,NC] = act( s1[m]*A1[M,K1] @ W1[NC,K1]^T  (+ A2[M,K2] @ W2[NC,K2]^T) + bias )
// Wave32 owns a 16-row x (16*NT)-col strip: A fragment built once per K-step
// (f32 load, optional row scale, cvt to f16), reused across NT WMMA tiles.
// A-fragment lane layout per CDNA5 ISA 7.12.2 (16-bit A 16x32); B mirrors it
// with the column index in lane[3:0].
template<int K1, int K2, int NC, int NT, int ACT>
__global__ void wmma_gemm_kernel(const float* __restrict__ A1, const float* __restrict__ S1,
                                 const float* __restrict__ W1,
                                 const float* __restrict__ A2, const float* __restrict__ W2,
                                 const float* __restrict__ bias,
                                 const float* __restrict__ prelu,
                                 float* __restrict__ C, int M) {
  const int lane = threadIdx.x & 31;
  const int wid  = blockIdx.x * (blockDim.x >> 5) + (threadIdx.x >> 5);
  constexpr int STRIPS = NC / (16 * NT);
  const int tilesM = (M + 15) >> 4;
  if (wid >= tilesM * STRIPS) return;
  const int tm  = wid / STRIPS;
  const int nc0 = (wid % STRIPS) * (16 * NT);
  const int lr = lane & 15;            // row (A) / col (B) owned by this lane
  const int kh = (lane >> 4) << 3;     // K half-offset (0 or 8)
  const long long ma  = (long long)(tm * 16 + lr);
  const int       nbl = nc0 + lr;      // base column for this lane's B loads

  v8f acc[NT] = {};
  const float s1 = S1 ? S1[ma] : 1.0f;

  for (int k0 = 0; k0 < K1; k0 += 32) {
    v16h a;
#pragma unroll
    for (int v = 0; v < 8; ++v) {
      const int kb = (v < 4) ? (kh + 2 * v) : (16 + kh + 2 * (v - 4));
      const int k = k0 + kb;
      a[2 * v]     = (_Float16)(A1[ma * K1 + k]     * s1);
      a[2 * v + 1] = (_Float16)(A1[ma * K1 + k + 1] * s1);
    }
#pragma unroll
    for (int t = 0; t < NT; ++t) {
      const long long nb = (long long)(nbl + 16 * t);
      v16h b;
#pragma unroll
      for (int v = 0; v < 8; ++v) {
        const int kb = (v < 4) ? (kh + 2 * v) : (16 + kh + 2 * (v - 4));
        const int k = k0 + kb;
        b[2 * v]     = (_Float16)W1[nb * K1 + k];
        b[2 * v + 1] = (_Float16)W1[nb * K1 + k + 1];
      }
      acc[t] = __builtin_amdgcn_wmma_f32_16x16x32_f16(false, a, false, b, (short)0, acc[t],
                                                      false, false);
    }
  }
  if constexpr (K2 > 0) {
    for (int k0 = 0; k0 < K2; k0 += 32) {
      v16h a;
#pragma unroll
      for (int v = 0; v < 8; ++v) {
        const int kb = (v < 4) ? (kh + 2 * v) : (16 + kh + 2 * (v - 4));
        const int k = k0 + kb;
        a[2 * v]     = (_Float16)A2[ma * K2 + k];
        a[2 * v + 1] = (_Float16)A2[ma * K2 + k + 1];
      }
#pragma unroll
      for (int t = 0; t < NT; ++t) {
        const long long nb = (long long)(nbl + 16 * t);
        v16h b;
#pragma unroll
        for (int v = 0; v < 8; ++v) {
          const int kb = (v < 4) ? (kh + 2 * v) : (16 + kh + 2 * (v - 4));
          const int k = k0 + kb;
          b[2 * v]     = (_Float16)W2[nb * K2 + k];
          b[2 * v + 1] = (_Float16)W2[nb * K2 + k + 1];
        }
        acc[t] = __builtin_amdgcn_wmma_f32_16x16x32_f16(false, a, false, b, (short)0, acc[t],
                                                        false, false);
      }
    }
  }

  float pa = 0.0f;
  if (ACT == 2) pa = prelu[0];
#pragma unroll
  for (int t = 0; t < NT; ++t) {
    const int nb = nbl + 16 * t;
    const float bv = bias ? bias[nb] : 0.0f;
#pragma unroll
    for (int r = 0; r < 8; ++r) {
      const int m = tm * 16 + r + ((lane >> 4) << 3);
      if (m < M) {
        float v = acc[t][r] + bv;
        if (ACT == 1) v = fmaxf(v, 0.0f);
        else if (ACT == 2) v = (v >= 0.0f) ? v : pa * v;
        C[(long long)m * NC + nb] = v;
      }
    }
  }
}

// --------------------------- elementwise nets ------------------------------
// gate = sigmoid(h[n,:] . w + b), wave per row, two elems per lane
__global__ void gate_kernel(const float* __restrict__ h, const float* __restrict__ w,
                            const float* __restrict__ b, float* __restrict__ gate) {
  int wv = (blockIdx.x * blockDim.x + threadIdx.x) >> 5;
  int lane = threadIdx.x & 31;
  if (wv >= NNODES) return;
  float s = h[(long long)wv * HDIM + lane] * w[lane]
          + h[(long long)wv * HDIM + 32 + lane] * w[32 + lane];
#pragma unroll
  for (int off = 16; off; off >>= 1) s += __shfl_xor(s, off, 32);
  if (lane == 0) gate[wv] = sigf(s + b[0]);
}

// GRU: h = (1-z)*n + z*h  (gi/gh already include their biases from the GEMM)
__global__ void gru_combine_kernel(const float* __restrict__ gi, const float* __restrict__ gh,
                                   float* __restrict__ h) {
  long long t = (long long)blockIdx.x * blockDim.x + threadIdx.x;
  if (t >= (long long)NNODES * HDIM) return;
  int n = (int)(t >> 6), f = (int)(t & 63);
  const float* gin = gi + (long long)n * 3 * HDIM;
  const float* ghn = gh + (long long)n * 3 * HDIM;
  float r  = sigf(gin[f] + ghn[f]);
  float z  = sigf(gin[HDIM + f] + ghn[HDIM + f]);
  float nn = tanhf(gin[2 * HDIM + f] + r * ghn[2 * HDIM + f]);
  h[t] = (1.0f - z) * nn + z * h[t];
}

__global__ void concat_kernel(const float* __restrict__ proj, const float* __restrict__ hs,
                              const float* __restrict__ hr, float* __restrict__ feats) {
  long long t = (long long)blockIdx.x * blockDim.x + threadIdx.x;
  if (t >= (long long)NNODES * DDIM) return;
  int n = (int)(t / DDIM), f = (int)(t % DDIM);
  float v = (f < HDIM) ? proj[(long long)n * HDIM + f]
          : (f < 2 * HDIM) ? hs[(long long)n * HDIM + f - HDIM]
                           : hr[(long long)n * HDIM + f - 2 * HDIM];
  feats[t] = v;
}

// LSTM combine (g already includes bih; add bhh here)
__global__ void lstm_combine_kernel(const float* __restrict__ g, const float* __restrict__ bhh,
                                    float* __restrict__ hl, float* __restrict__ cl) {
  int t = blockIdx.x * blockDim.x + threadIdx.x;
  if (t >= BGRAPH * DDIM) return;
  int b = t / DDIM, f = t % DDIM;
  const float* gb = g + (long long)b * 4 * DDIM;
  float i  = gb[f]            + bhh[f];
  float ff = gb[DDIM + f]     + bhh[DDIM + f];
  float gg = gb[2 * DDIM + f] + bhh[2 * DDIM + f];
  float o  = gb[3 * DDIM + f] + bhh[3 * DDIM + f];
  float cn = sigf(ff) * cl[t] + sigf(i) * tanhf(gg);
  cl[t] = cn;
  hl[t] = sigf(o) * tanhf(cn);
}

// e[n] = feats[n,:] . hl[batch[n],:]   (wave per node, 6 elems per lane)
__global__ void attn_dot_kernel(const float* __restrict__ feats, const float* __restrict__ hl,
                                const int* __restrict__ batch, float* __restrict__ e) {
  int wv = (blockIdx.x * blockDim.x + threadIdx.x) >> 5;
  int lane = threadIdx.x & 31;
  if (wv >= NNODES) return;
  const float* fr = feats + (long long)wv * DDIM;
  const float* qr = hl + (long long)batch[wv] * DDIM;
  float s = 0.0f;
#pragma unroll
  for (int j = 0; j < 6; ++j) s += fr[lane + 32 * j] * qr[lane + 32 * j];
#pragma unroll
  for (int off = 16; off; off >>= 1) s += __shfl_xor(s, off, 32);
  if (lane == 0) e[wv] = s;
}

__global__ void segmax_kernel(const float* __restrict__ e, const int* __restrict__ batch,
                              unsigned* __restrict__ mmax) {
  int i = blockIdx.x * blockDim.x + threadIdx.x;
  if (i < NNODES) atomicMax(&mmax[batch[i]], f2ord(e[i]));
}

__global__ void expsum_kernel(const float* __restrict__ e, const int* __restrict__ batch,
                              const unsigned* __restrict__ mmax, float* __restrict__ ex,
                              float* __restrict__ denom) {
  int i = blockIdx.x * blockDim.x + threadIdx.x;
  if (i >= NNODES) return;
  int b = batch[i];
  float v = __expf(e[i] - ord2f(mmax[b]));
  ex[i] = v;
  atomicAdd(&denom[b], v);
}

__global__ void racc_kernel(const float* __restrict__ feats, const float* __restrict__ ex,
                            const float* __restrict__ denom, const int* __restrict__ batch,
                            float* __restrict__ rr) {
  long long t = (long long)blockIdx.x * blockDim.x + threadIdx.x;
  if (t >= (long long)NNODES * DDIM) return;
  int n = (int)(t / DDIM), f = (int)(t % DDIM);
  int b = batch[n];
  atomicAdd(&rr[(long long)b * DDIM + f], (ex[n] / denom[b]) * feats[t]);
}

__global__ void qstar_kernel(const float* __restrict__ hl, const float* __restrict__ rr,
                             float* __restrict__ qs) {
  int t = blockIdx.x * blockDim.x + threadIdx.x;
  if (t >= BGRAPH * 2 * DDIM) return;
  int b = t / (2 * DDIM), f = t % (2 * DDIM);
  qs[t] = (f < DDIM) ? hl[(long long)b * DDIM + f] : rr[(long long)b * DDIM + (f - DDIM)];
}

// ---------------------------------------------------------------------------
extern "C" void kernel_launch(void* const* d_in, const int* in_sizes, int n_in,
                              void* d_out, int out_size, void* d_ws, size_t ws_size,
                              hipStream_t stream) {
  (void)in_sizes; (void)n_in; (void)out_size; (void)ws_size;

  const float* x        = (const float*)d_in[0];
  const int*   ei       = (const int*)d_in[1];
  const int*   src      = ei;
  const int*   dst      = ei + NEDGES;
  const int*   batch    = (const int*)d_in[3];
  const float* W_p      = (const float*)d_in[4];
  const float* b_p      = (const float*)d_in[5];
  const float* W_l      = (const float*)d_in[6];
  const float* b_l      = (const float*)d_in[7];
  const float* W_r      = (const float*)d_in[8];
  const float* gs_W     = (const float*)d_in[9];
  const float* gs_b     = (const float*)d_in[10];
  const float* gr_W     = (const float*)d_in[11];
  const float* gr_b     = (const float*)d_in[12];
  const float* grus_Wih = (const float*)d_in[13];
  const float* grus_Whh = (const float*)d_in[14];
  const float* grus_bih = (const float*)d_in[15];
  const float* grus_bhh = (const float*)d_in[16];
  const float* grur_Wih = (const float*)d_in[17];
  const float* grur_Whh = (const float*)d_in[18];
  const float* grur_bih = (const float*)d_in[19];
  const float* grur_bhh = (const float*)d_in[20];
  const float* lstm_Wih = (const float*)d_in[21];
  const float* lstm_Whh = (const float*)d_in[22];
  const float* lstm_bih = (const float*)d_in[23];
  const float* lstm_bhh = (const float*)d_in[24];
  const float* W_sp     = (const float*)d_in[25];
  const float* b_sp     = (const float*)d_in[26];
  const float* prelu_a  = (const float*)d_in[27];

  float* out       = (float*)d_out;                       // [B, 1024]
  float* gates_out = out + (long long)BGRAPH * ROUT;      // [3, N, 1]

  // ---- workspace carve (floats). Peak ~ (2 + 3*64 + 512)*N + small*B -----
  float* ws = (float*)d_ws;
  long long o = 0;
  float* recip = ws + o; o += NNODES;
  float* gateS = ws + o; o += NNODES;
  float* proj  = ws + o; o += (long long)NNODES * HDIM;
  float* hS    = ws + o; o += (long long)NNODES * HDIM;
  float* hR    = ws + o; o += (long long)NNODES * HDIM;
  float* big   = ws + o; o += (long long)NNODES * 512;    // agg|am|gi|gh; reused as feats|e|ex
  float* agg = big;
  float* am  = big + (long long)NNODES * HDIM;
  float* gi  = big + (long long)NNODES * 2 * HDIM;
  float* gh  = big + (long long)NNODES * 5 * HDIM;
  float* feats = big;                                     // [N,192] overlays agg/am/gi(part)
  float* e_buf = big + (long long)NNODES * DDIM;
  float* ex_buf = e_buf + NNODES;
  float* qs    = ws + o; o += (long long)BGRAPH * 2 * DDIM;
  float* hl    = ws + o; o += (long long)BGRAPH * DDIM;
  float* cl    = ws + o; o += (long long)BGRAPH * DDIM;
  float* gbuf  = ws + o; o += (long long)BGRAPH * 4 * DDIM;
  unsigned* mmaxu = (unsigned*)(ws + o); o += BGRAPH;
  float* denom = ws + o; o += BGRAPH;
  float* rr    = ws + o; o += (long long)BGRAPH * DDIM;

  const int TB = 256;
#define GRID1(n) ((unsigned)(((long long)(n) + TB - 1) / TB))
#define FILL(p, n) fill_zero_f32<<<GRID1(n), TB, 0, stream>>>((p), (long long)(n))
// waves = ceil(M/16) * (NC/(16*NT)); 8 waves per 256-thread block
#define GEMM_GRID(M, NC, NT) \
  ((unsigned)(((((M) + 15) / 16) * ((NC) / (16 * (NT))) + 7) / 8))

  // ---- degree -> reciprocal ----
  FILL(recip, NNODES);
  deg_kernel<<<GRID1(NEDGES), TB, 0, stream>>>(dst, recip);
  recip_deg_kernel<<<GRID1(NNODES), TB, 0, stream>>>(recip);

  // ---- proj = relu(x @ W_p^T + b_p); h_s = h_r = proj ----
  wmma_gemm_kernel<128, 0, 64, 4, 1><<<GEMM_GRID(NNODES, 64, 4), TB, 0, stream>>>(
      x, nullptr, W_p, nullptr, nullptr, b_p, nullptr, proj, NNODES);
  hipMemcpyAsync(hS, proj, (size_t)NNODES * HDIM * sizeof(float), hipMemcpyDeviceToDevice, stream);
  hipMemcpyAsync(hR, proj, (size_t)NNODES * HDIM * sizeof(float), hipMemcpyDeviceToDevice, stream);

  // ---- message passing ----
  for (int step = 0; step < 3; ++step) {
    FILL(agg, (long long)NNODES * HDIM);
    edge_agg_kernel<<<GRID1((long long)NEDGES * HDIM), TB, 0, stream>>>(src, dst, hS, agg);
    // am = relu((agg/deg) @ W_l^T + b_l + h_s @ W_r^T)   (1/deg fused as A1 row scale)
    wmma_gemm_kernel<64, 64, 64, 4, 1><<<GEMM_GRID(NNODES, 64, 4), TB, 0, stream>>>(
        agg, recip, W_l, hS, W_r, b_l, nullptr, am, NNODES);

    // --- s-branch GRU ---
    gate_kernel<<<GRID1((long long)NNODES * 32), TB, 0, stream>>>(hS, gs_W, gs_b, gateS);
    wmma_gemm_kernel<64, 0, 192, 6, 0><<<GEMM_GRID(NNODES, 192, 6), TB, 0, stream>>>(
        am, gateS, grus_Wih, nullptr, nullptr, grus_bih, nullptr, gi, NNODES);
    wmma_gemm_kernel<64, 0, 192, 6, 0><<<GEMM_GRID(NNODES, 192, 6), TB, 0, stream>>>(
        hS, nullptr, grus_Whh, nullptr, nullptr, grus_bhh, nullptr, gh, NNODES);
    gru_combine_kernel<<<GRID1((long long)NNODES * HDIM), TB, 0, stream>>>(gi, gh, hS);

    // --- r-branch GRU (gate_r written straight into output slot 2) ---
    float* gateR = gates_out + (long long)step * NNODES;
    gate_kernel<<<GRID1((long long)NNODES * 32), TB, 0, stream>>>(hR, gr_W, gr_b, gateR);
    wmma_gemm_kernel<64, 0, 192, 6, 0><<<GEMM_GRID(NNODES, 192, 6), TB, 0, stream>>>(
        am, gateR, grur_Wih, nullptr, nullptr, grur_bih, nullptr, gi, NNODES);
    wmma_gemm_kernel<64, 0, 192, 6, 0><<<GEMM_GRID(NNODES, 192, 6), TB, 0, stream>>>(
        hR, nullptr, grur_Whh, nullptr, nullptr, grur_bhh, nullptr, gh, NNODES);
    gru_combine_kernel<<<GRID1((long long)NNODES * HDIM), TB, 0, stream>>>(gi, gh, hR);
  }

  // ---- feats = [proj | h_s | h_r]  (overlays agg/am/gi region) ----
  concat_kernel<<<GRID1((long long)NNODES * DDIM), TB, 0, stream>>>(proj, hS, hR, feats);

  // ---- Set2Set ----
  FILL(qs, (long long)BGRAPH * 2 * DDIM);
  FILL(hl, (long long)BGRAPH * DDIM);
  FILL(cl, (long long)BGRAPH * DDIM);
  for (int step = 0; step < 3; ++step) {
    // g = q_star @ Wih^T + bih + hl @ Whh^T   (bhh added in combine)
    wmma_gemm_kernel<384, 192, 768, 4, 0><<<GEMM_GRID(BGRAPH, 768, 4), TB, 0, stream>>>(
        qs, nullptr, lstm_Wih, hl, lstm_Whh, lstm_bih, nullptr, gbuf, BGRAPH);
    lstm_combine_kernel<<<GRID1(BGRAPH * DDIM), TB, 0, stream>>>(gbuf, lstm_bhh, hl, cl);

    FILL((float*)mmaxu, BGRAPH);     // ordered-uint 0 == lowest
    FILL(denom, BGRAPH);
    FILL(rr, (long long)BGRAPH * DDIM);
    attn_dot_kernel<<<GRID1((long long)NNODES * 32), TB, 0, stream>>>(feats, hl, batch, e_buf);
    segmax_kernel<<<GRID1(NNODES), TB, 0, stream>>>(e_buf, batch, mmaxu);
    expsum_kernel<<<GRID1(NNODES), TB, 0, stream>>>(e_buf, batch, mmaxu, ex_buf, denom);
    racc_kernel<<<GRID1((long long)NNODES * DDIM), TB, 0, stream>>>(feats, ex_buf, denom, batch, rr);
    qstar_kernel<<<GRID1(BGRAPH * 2 * DDIM), TB, 0, stream>>>(hl, rr, qs);
  }

  // ---- sparsify: out = PReLU(q_star @ W_sp^T + b_sp) ----
  wmma_gemm_kernel<384, 0, 1024, 4, 2><<<GEMM_GRID(BGRAPH, 1024, 4), TB, 0, stream>>>(
      qs, nullptr, W_sp, nullptr, nullptr, b_sp, prelu_a, out, BGRAPH);

#undef GRID1
#undef FILL
#undef GEMM_GRID
}
